// TransformerEncoderLayer_88948772700323
// MI455X (gfx1250) — matrix operand order repaired and checked by reading, not compiled, not run on hardware
//
#include <hip/hip_runtime.h>
#include <stdint.h>

#define D_MODEL 1024
#define D_FF    4096
#define NSEQ    2048
#define NBATCH  4
#define NHEADS  16
#define HDIM    64
#define MTOK    (NSEQ * NBATCH)   // 8192 token rows
#define C3      (3 * D_MODEL)     // 3072

typedef __attribute__((ext_vector_type(16))) __bf16 v16bf;
typedef __attribute__((ext_vector_type(8)))  float  v8f;

typedef unsigned short u16;
typedef unsigned int   u32;

__device__ __forceinline__ u16 f2bf(float f) {
  u32 u = __float_as_uint(f);
  u += 0x7fffu + ((u >> 16) & 1u);   // round-to-nearest-even
  return (u16)(u >> 16);
}

union Frag { u32 u[8]; v16bf v; };

__device__ __forceinline__ v8f vzero8() {
  v8f z;
#pragma unroll
  for (int i = 0; i < 8; ++i) z[i] = 0.0f;
  return z;
}

// A-fragment (16x32 bf16). ISA layout: lanes 0-15 row M=lane hold K={0..7,16..23},
// lanes 16-31 row M=lane-16 hold K={8..15,24..31}; 2 bf16 per VGPR.
__device__ __forceinline__ void load_afrag(const u16* __restrict__ base, int ld,
                                           int row0, int kbase, Frag& f) {
  const int lane = threadIdx.x & 31;
  const int half = lane >> 4, r = lane & 15;
  const u16* p = base + (size_t)(row0 + r) * (size_t)ld + kbase + half * 8;
#pragma unroll
  for (int i = 0; i < 4; ++i) {
    f.u[i]     = *(const u32*)(p + 2 * i);
    f.u[i + 4] = *(const u32*)(p + 16 + 2 * i);
  }
}

// B-fragment (32x16 bf16) gathered from "B-transposed" memory (row n contiguous
// in K). ISA layout: lanes 0-15 col N=lane hold K={0..7,16..23}, lanes 16-31 hold K={8..15,24..31}.
__device__ __forceinline__ void load_bfrag(const u16* __restrict__ base, int ld,
                                           int col0, int kbase, Frag& f) {
  const int lane = threadIdx.x & 31;
  const int half = lane >> 4, r = lane & 15;
  const u16* p = base + (size_t)(col0 + r) * (size_t)ld + kbase + half * 8;
#pragma unroll
  for (int i = 0; i < 4; ++i) {
    f.u[i]     = *(const u32*)(p + 2 * i);
    f.u[i + 4] = *(const u32*)(p + 16 + 2 * i);
  }
}

__device__ __forceinline__ v8f wmma_bf16(const Frag& a, const Frag& b, v8f c) {
  return __builtin_amdgcn_wmma_f32_16x16x32_bf16(false, a.v, false, b.v,
                                                 (short)0, c, false, false);
}

// ---------------------------------------------------------------------------
// LayerNorm (fp32 in, bf16 out), one 256-thread block per token row
// ---------------------------------------------------------------------------
__global__ __launch_bounds__(256)
void layernorm_bf16(const float* __restrict__ src, const float* __restrict__ g,
                    const float* __restrict__ b, u16* __restrict__ xn) {
  __shared__ float s1[256], s2[256];
  const int row = blockIdx.x;
  const int tid = threadIdx.x;
  const float* x = src + (size_t)row * D_MODEL;
  float a = 0.f, aa = 0.f;
  for (int c = tid; c < D_MODEL; c += 256) { float v = x[c]; a += v; aa += v * v; }
  s1[tid] = a; s2[tid] = aa;
  __syncthreads();
  for (int s = 128; s > 0; s >>= 1) {
    if (tid < s) { s1[tid] += s1[tid + s]; s2[tid] += s2[tid + s]; }
    __syncthreads();
  }
  const float mean = s1[0] * (1.0f / D_MODEL);
  const float var  = s2[0] * (1.0f / D_MODEL) - mean * mean;
  const float rstd = rsqrtf(var + 1e-5f);
  u16* o = xn + (size_t)row * D_MODEL;
  for (int c = tid; c < D_MODEL; c += 256)
    o[c] = f2bf((x[c] - mean) * rstd * g[c] + b[c]);
}

// ---------------------------------------------------------------------------
// Weight transpose + fp32 -> bf16: out[n*K + k] = in[k*N + n]
// ---------------------------------------------------------------------------
__global__ __launch_bounds__(256)
void transpose_f32_to_bf16(const float* __restrict__ in, u16* __restrict__ out,
                           int K, int N) {
  const size_t idx = (size_t)blockIdx.x * 256 + threadIdx.x;
  const int n = (int)(idx / (size_t)K);
  const int k = (int)(idx - (size_t)n * K);
  out[idx] = f2bf(in[(size_t)k * N + n]);
}

// V^T per head: vt[b,h,d,n] = qkv[n, b, V-part, h, d]
__global__ __launch_bounds__(256)
void build_vt(const u16* __restrict__ qkv, u16* __restrict__ vt) {
  const size_t idx = (size_t)blockIdx.x * 256 + threadIdx.x;
  const int n = (int)(idx & (NSEQ - 1));
  size_t t = idx >> 11;
  const int d = (int)(t & (HDIM - 1)); t >>= 6;
  const int h = (int)(t & (NHEADS - 1));
  const int b = (int)(t >> 4);
  vt[idx] = qkv[((size_t)n * NBATCH + b) * C3 + 2 * D_MODEL + h * HDIM + d];
}

// ---------------------------------------------------------------------------
// Generic bf16 WMMA GEMM: one wave computes a 64x64 f32 tile (4x4 sub-tiles).
// Per k-step: 16 b128 loads feed 16 WMMAs (512 B/WMMA of cache traffic).
// A: (M x K) bf16 row-major.  BT: (N x K) bf16 row-major (i.e. B transposed).
// ---------------------------------------------------------------------------
#define GM_MODE_QKV  0  // -> outB bf16, no bias
#define GM_MODE_RES  1  // acc+bias+res -> outF f32 AND outB bf16
#define GM_MODE_GELU 2  // gelu(acc+bias) -> outB bf16
#define GM_MODE_OUT  3  // acc+bias+res -> outF f32 only

__global__ __launch_bounds__(32)
void gemm_bf16_wmma(const u16* __restrict__ A, const u16* __restrict__ BT,
                    int K, int NN,
                    const float* __restrict__ bias,
                    const float* __restrict__ res,
                    float* __restrict__ outF, u16* __restrict__ outB, int mode) {
  const int mb = blockIdx.x * 64;
  const int nb = blockIdx.y * 64;
  const int lane = threadIdx.x & 31;
  const int half = lane >> 4, r = lane & 15;

  v8f acc[4][4];
#pragma unroll
  for (int i = 0; i < 4; ++i)
#pragma unroll
    for (int j = 0; j < 4; ++j) acc[i][j] = vzero8();

  for (int kb = 0; kb < K; kb += 32) {
    Frag af[4], bf[4];
#pragma unroll
    for (int i = 0; i < 4; ++i) load_afrag(A, K, mb + i * 16, kb, af[i]);
#pragma unroll
    for (int j = 0; j < 4; ++j) load_bfrag(BT, K, nb + j * 16, kb, bf[j]);
#pragma unroll
    for (int i = 0; i < 4; ++i)
#pragma unroll
      for (int j = 0; j < 4; ++j)
        acc[i][j] = wmma_bf16(af[i], bf[j], acc[i][j]);
  }

#pragma unroll
  for (int mi = 0; mi < 4; ++mi) {
#pragma unroll
    for (int j = 0; j < 4; ++j) {
      const int col = nb + j * 16 + r;
      const float bv = bias ? bias[col] : 0.0f;
#pragma unroll
      for (int i = 0; i < 8; ++i) {
        const int row = mb + mi * 16 + half * 8 + i;
        const size_t off = (size_t)row * (size_t)NN + col;
        float v = acc[mi][j][i] + bv;
        if (mode == GM_MODE_QKV) {
          outB[off] = f2bf(v);
        } else if (mode == GM_MODE_RES) {
          v += res[off];
          outF[off] = v;
          outB[off] = f2bf(v);
        } else if (mode == GM_MODE_GELU) {
          const float gl = 0.5f * v * (1.0f + erff(v * 0.70710678118654752f));
          outB[off] = f2bf(gl);
        } else {
          outF[off] = v + res[off];
        }
      }
    }
  }
}

// ---------------------------------------------------------------------------
// Flash attention: one block (4 waves) per (b, h, 16-query tile).
// Keys processed in chunks of 512 with online softmax; all matmuls via WMMA.
// Softmax stats are parallelized: 8 threads per query row with LDS tree-reduce.
// ---------------------------------------------------------------------------
__global__ __launch_bounds__(128)
void flash_attn(const u16* __restrict__ qkv, const u16* __restrict__ vt,
                u16* __restrict__ oB) {
  __shared__ float S[16][512];
  __shared__ u16   P[16][512];
  __shared__ float mS[16], lS[16], scS[16];
  __shared__ float red[16][8];

  const int tid = threadIdx.x;
  const int wave = tid >> 5;
  const int lane = tid & 31;
  const int half = lane >> 4, r = lane & 15;
  const int srow = tid >> 3;        // stats: row handled by this thread
  const int ssub = tid & 7;         // stats: 8-way split of the 512 columns

  int bid = blockIdx.x;
  const int qt = bid & 127; bid >>= 7;
  const int h = bid & 15;
  const int b = bid >> 4;

  const int ldq = NBATCH * C3;  // row stride of Q/K over sequence index n
  const u16* qbase = qkv + (size_t)b * C3 + h * HDIM;
  const u16* kbase = qkv + (size_t)b * C3 + D_MODEL + h * HDIM;
  const u16* vbase = vt + (size_t)(b * NHEADS + h) * HDIM * NSEQ;

  Frag aq0, aq1;                       // Q tile fragments, reused all chunks
  load_afrag(qbase, ldq, qt * 16, 0, aq0);
  load_afrag(qbase, ldq, qt * 16, 32, aq1);

  if (tid < 16) { mS[tid] = -1e30f; lS[tid] = 0.0f; }
  v8f acc = vzero8();
  __syncthreads();

  for (int kc = 0; kc < NSEQ; kc += 512) {
    // ---- S chunk = Q K^T * scale; each wave covers 128 keys (8 tiles) ----
    for (int t = 0; t < 8; ++t) {
      const int key0 = kc + wave * 128 + t * 16;
      Frag bk0, bk1;
      load_bfrag(kbase, ldq, key0, 0, bk0);
      load_bfrag(kbase, ldq, key0, 32, bk1);
      v8f s = vzero8();
      s = wmma_bf16(aq0, bk0, s);
      s = wmma_bf16(aq1, bk1, s);
      const int cb = wave * 128 + t * 16 + r;
#pragma unroll
      for (int i = 0; i < 8; ++i) S[half * 8 + i][cb] = s[i] * 0.125f;  // hd^-0.5
    }
    __syncthreads();

    // ---- online softmax stats: 8 threads per row ----
    {
      float pm = -1e30f;
      const int c0 = ssub * 64;
#pragma unroll 8
      for (int c = 0; c < 64; ++c) pm = fmaxf(pm, S[srow][c0 + c]);
      red[srow][ssub] = pm;
    }
    __syncthreads();
    {
      float cm = red[srow][0];
#pragma unroll
      for (int i = 1; i < 8; ++i) cm = fmaxf(cm, red[srow][i]);
      const float mold = mS[srow];           // read-before-update (barrier below)
      const float mnew = fmaxf(mold, cm);
      float psum = 0.0f;
      const int c0 = ssub * 64;
#pragma unroll 8
      for (int c = 0; c < 64; ++c) {
        const float e = __expf(S[srow][c0 + c] - mnew);
        P[srow][c0 + c] = f2bf(e);
        psum += e;
      }
      red[srow][ssub] = psum;
      __syncthreads();
      if (ssub == 0) {
        float sum = red[srow][0];
#pragma unroll
        for (int i = 1; i < 8; ++i) sum += red[srow][i];
        const float sc = __expf(mold - mnew);
        scS[srow] = sc;
        lS[srow] = lS[srow] * sc + sum;
        mS[srow] = mnew;
      }
    }
    __syncthreads();

    // ---- rescale running accumulator, then O += P V ----
#pragma unroll
    for (int i = 0; i < 8; ++i) acc[i] *= scS[half * 8 + i];

    for (int kk = 0; kk < 16; ++kk) {
      Frag ap, bv;
      {  // A-fragment gather from LDS-resident P (row-major 16x512)
        const u16* p = &P[r][kk * 32 + half * 8];
#pragma unroll
        for (int i = 0; i < 4; ++i) {
          ap.u[i]     = *(const u32*)(p + 2 * i);
          ap.u[i + 4] = *(const u32*)(p + 16 + 2 * i);
        }
      }
      load_bfrag(vbase, NSEQ, wave * 16, kc + kk * 32, bv);  // from V^T
      acc = wmma_bf16(ap, bv, acc);
    }
    __syncthreads();
  }

  // ---- normalize + store with the reference's (N,H,B,hd)->(N,B,C) mapping ----
  const int d = wave * 16 + r;
#pragma unroll
  for (int i = 0; i < 8; ++i) {
    const int n = qt * 16 + half * 8 + i;
    const float v = acc[i] / lS[half * 8 + i];
    const size_t off = ((size_t)n * NBATCH + (h >> 2)) * D_MODEL
                     + (h & 3) * 256 + b * HDIM + d;
    oB[off] = f2bf(v);
  }
}

// ---------------------------------------------------------------------------
// Host launcher
// ---------------------------------------------------------------------------
extern "C" void kernel_launch(void* const* d_in, const int* in_sizes, int n_in,
                              void* d_out, int out_size, void* d_ws, size_t ws_size,
                              hipStream_t stream) {
  const float* src    = (const float*)d_in[0];
  const float* ln_g   = (const float*)d_in[1];
  const float* ln_b   = (const float*)d_in[2];
  const float* w_qkv  = (const float*)d_in[3];
  const float* w_proj = (const float*)d_in[4];
  const float* b_proj = (const float*)d_in[5];
  const float* w1     = (const float*)d_in[6];
  const float* b1     = (const float*)d_in[7];
  const float* w2     = (const float*)d_in[8];
  const float* b2     = (const float*)d_in[9];
  float* out = (float*)d_out;
  (void)in_sizes; (void)n_in; (void)out_size; (void)ws_size;

  char* p = (char*)d_ws;
  auto carve = [&](size_t bytes) {
    char* q = p;
    p += (bytes + 255) & ~(size_t)255;
    return q;
  };

  u16*   xnB    = (u16*)carve((size_t)MTOK * D_MODEL * 2);          // LN output
  u16*   wqkvT  = (u16*)carve((size_t)C3 * D_MODEL * 2);            // w_qkv^T
  u16*   wprojT = (u16*)carve((size_t)D_MODEL * D_MODEL * 2);       // w_proj^T
  u16*   w1T    = (u16*)carve((size_t)D_FF * D_MODEL * 2);          // w1^T
  u16*   w2T    = (u16*)carve((size_t)D_MODEL * D_FF * 2);          // w2^T
  u16*   qkvB   = (u16*)carve((size_t)MTOK * C3 * 2);               // qkv bf16
  u16*   vtB    = (u16*)carve((size_t)NBATCH * NHEADS * HDIM * NSEQ * 2);
  u16*   oBuf   = (u16*)carve((size_t)MTOK * D_MODEL * 2);          // attn out bf16
  float* s2F    = (float*)carve((size_t)MTOK * D_MODEL * 4);        // residual f32
  u16*   s2B    = (u16*)carve((size_t)MTOK * D_MODEL * 2);          // residual bf16
  u16*   hB     = (u16*)carve((size_t)MTOK * D_FF * 2);             // gelu out bf16

  layernorm_bf16<<<MTOK, 256, 0, stream>>>(src, ln_g, ln_b, xnB);

  transpose_f32_to_bf16<<<(C3 * D_MODEL) / 256, 256, 0, stream>>>(w_qkv, wqkvT, D_MODEL, C3);
  transpose_f32_to_bf16<<<(D_MODEL * D_MODEL) / 256, 256, 0, stream>>>(w_proj, wprojT, D_MODEL, D_MODEL);
  transpose_f32_to_bf16<<<(D_FF * D_MODEL) / 256, 256, 0, stream>>>(w1, w1T, D_MODEL, D_FF);
  transpose_f32_to_bf16<<<(D_MODEL * D_FF) / 256, 256, 0, stream>>>(w2, w2T, D_FF, D_MODEL);

  // QKV projection (no bias) -> bf16
  gemm_bf16_wmma<<<dim3(MTOK / 64, C3 / 64), 32, 0, stream>>>(
      xnB, wqkvT, D_MODEL, C3, nullptr, nullptr, nullptr, qkvB, GM_MODE_QKV);

  build_vt<<<(NBATCH * NHEADS * HDIM * NSEQ) / 256, 256, 0, stream>>>(qkvB, vtB);

  flash_attn<<<NBATCH * NHEADS * (NSEQ / 16), 128, 0, stream>>>(qkvB, vtB, oBuf);

  // out-proj + bias + residual(src) -> f32 + bf16
  gemm_bf16_wmma<<<dim3(MTOK / 64, D_MODEL / 64), 32, 0, stream>>>(
      oBuf, wprojT, D_MODEL, D_MODEL, b_proj, src, s2F, s2B, GM_MODE_RES);

  // FF1 + bias + exact GELU -> bf16
  gemm_bf16_wmma<<<dim3(MTOK / 64, D_FF / 64), 32, 0, stream>>>(
      s2B, w1T, D_MODEL, D_FF, b1, nullptr, nullptr, hB, GM_MODE_GELU);

  // FF2 + bias + residual -> final f32 output
  gemm_bf16_wmma<<<dim3(MTOK / 64, D_MODEL / 64), 32, 0, stream>>>(
      hB, w2T, D_FF, D_MODEL, b2, s2F, out, nullptr, GM_MODE_OUT);
}
